// SpellSeq2Seq_11613591569134
// MI455X (gfx1250) — compile-verified, matrix operand order checked by reading, not run on hardware
//
#include <hip/hip_runtime.h>
#include <hip/hip_bf16.h>
#include <stdint.h>

// Problem constants (from reference): B=128, S=T=256, V=512, E=256, H=512
#define B_ 128
#define S_ 256
#define T_ 256
#define V_ 512
#define E_ 256
#define H_ 512
#define NB 32     // persistent grid blocks (must all be resident)
#define NT 256    // threads per block = 8 waves
#define LPAD 8    // LDS row padding (elements) to spread banks

typedef __attribute__((ext_vector_type(16))) __bf16 v16bf;
typedef __attribute__((ext_vector_type(8)))  float  v8f;

__device__ __forceinline__ float sigm(float x) { return 1.0f / (1.0f + __expf(-x)); }

// ---------------------------------------------------------------------------
// CDNA5 async global->LDS copy (tracked by ASYNCcnt).  Each lane copies 16B
// from its global address to its LDS byte address.
// ---------------------------------------------------------------------------
__device__ __forceinline__ void async_copy_b128(unsigned lds_byte_addr, const void* gaddr) {
  asm volatile("global_load_async_to_lds_b128 %0, %1, off"
               :: "v"(lds_byte_addr), "v"(gaddr) : "memory");
}
__device__ __forceinline__ void wait_asynccnt0() {
  asm volatile("s_wait_asynccnt 0x0" ::: "memory");
}
__device__ __forceinline__ unsigned lds_addr_of(const void* p) {
  return (unsigned)(uintptr_t)p;   // low 32 bits of a flat LDS address = LDS offset
}

// ---------------------------------------------------------------------------
// WMMA fragment load: 16x32 bf16 tile.  Per CDNA5 ISA 16-bit A layout:
// lanes 0-15 hold row m=lane, K={k0..k0+7, k0+16..k0+23}; lanes 16-31 hold
// row m=lane-16, K={k0+8..k0+15, k0+24..k0+31} => two 16B chunks per lane.
// Works for global (global_load_b128) or LDS (ds_load_b128) base pointers.
// ---------------------------------------------------------------------------
__device__ __forceinline__ v16bf load_frag(const uint16_t* base, int ld, int row0, int k0) {
  const int lane = threadIdx.x & 31;
  const uint16_t* p = base + (size_t)(row0 + (lane & 15)) * (size_t)ld
                           + (size_t)(k0 + ((lane & 16) ? 8 : 0));
  union { v16bf v; uint4 u[2]; } f;
  f.u[0] = *reinterpret_cast<const uint4*>(p);
  f.u[1] = *reinterpret_cast<const uint4*>(p + 16);
  return f.v;
}

__device__ __forceinline__ v8f wmma_bf16(v16bf a, v16bf b, v8f c) {
  return __builtin_amdgcn_wmma_f32_16x16x32_bf16(false, a, false, b, (short)0, c, false, false);
}

// ---------------------------------------------------------------------------
// Grid-wide barrier for persistent kernels (monotonic arrival counter).
// ---------------------------------------------------------------------------
__device__ __forceinline__ void gsync(unsigned* cnt, unsigned& bar, unsigned nb) {
  __syncthreads();
  if (threadIdx.x == 0) {
    __threadfence();
    atomicAdd(cnt, 1u);
    ++bar;
    const unsigned target = nb * bar;
    while (*((volatile unsigned*)cnt) < target) __builtin_amdgcn_s_sleep(1);
    __threadfence();
  }
  __syncthreads();
}

// ---------------------------------------------------------------------------
// One wave: 16x16 GEMM tile -> fp32 out.  W rows start at wrow0 (W may be an
// LDS tile with local row numbering); n0 is the logical output column base.
// ---------------------------------------------------------------------------
__device__ __forceinline__ void gemm_out_tile(
    const uint16_t* A, int lda, int row0,
    const uint16_t* W, int ldw, int wrow0, int n0, int K,
    const float* __restrict__ bias,
    float* __restrict__ out, long long obase, long long orow) {
  v8f acc = {};
  for (int k = 0; k < K; k += 32) {
    v16bf a = load_frag(A, lda, row0, k);
    v16bf b = load_frag(W, ldw, wrow0, k);
    acc = wmma_bf16(a, b, acc);
  }
  const int lane = threadIdx.x & 31;
  const int n = n0 + (lane & 15);
  const int rofs = (lane & 16) ? 8 : 0;
  const float bv = bias[n];
  #pragma unroll
  for (int r = 0; r < 8; ++r) {
    const int m = row0 + rofs + r;
    out[obase + (long long)m * orow + n] = acc[r] + bv;
  }
}

// ---------------------------------------------------------------------------
// One wave: fused 4-gate LSTM tile.  Gate g's weight rows start at wr[g] in W
// (LDS tile: 0/16/32/48; global: g*512+j0).  Updates h,c (fp32, ld 512) and
// an optional bf16 mirror of h at obf[obase + b*ostride + n].
// ---------------------------------------------------------------------------
__device__ __forceinline__ void lstm_cell_tile(
    const uint16_t* A, int lda, int row0,
    const uint16_t* W, int ldw, int wr0, int wr1, int wr2, int wr3, int j0,
    int K,
    const float* __restrict__ bih, const float* __restrict__ bhh,
    float* __restrict__ h, float* __restrict__ c,
    __hip_bfloat16* obf, long long obase, long long ostride) {
  v8f gi = {}, gf = {}, gg = {}, go = {};
  for (int k = 0; k < K; k += 32) {
    v16bf a  = load_frag(A, lda, row0, k);          // shared across 4 gates
    v16bf wi = load_frag(W, ldw, wr0, k);
    v16bf wf = load_frag(W, ldw, wr1, k);
    v16bf wg = load_frag(W, ldw, wr2, k);
    v16bf wo = load_frag(W, ldw, wr3, k);
    gi = wmma_bf16(a, wi, gi);
    gf = wmma_bf16(a, wf, gf);
    gg = wmma_bf16(a, wg, gg);
    go = wmma_bf16(a, wo, go);
  }
  const int lane = threadIdx.x & 31;
  const int n = j0 + (lane & 15);
  const int rofs = (lane & 16) ? 8 : 0;
  const float bi = bih[n]          + bhh[n];
  const float bf = bih[512 + n]    + bhh[512 + n];
  const float bg = bih[1024 + n]   + bhh[1024 + n];
  const float bo = bih[1536 + n]   + bhh[1536 + n];
  #pragma unroll
  for (int r = 0; r < 8; ++r) {
    const int b = row0 + rofs + r;          // batch row
    const int idx = b * H_ + n;
    const float iv = sigm(gi[r] + bi);
    const float fv = sigm(gf[r] + bf);
    const float gv = tanhf(gg[r] + bg);
    const float ov = sigm(go[r] + bo);
    const float cn = fv * c[idx] + iv * gv;
    const float hn = ov * tanhf(cn);
    c[idx] = cn;
    h[idx] = hn;
    if (obf) obf[obase + (long long)b * ostride + n] = __float2bfloat16(hn);
  }
}

// ---------------------------------------------------------------------------
// Setup kernels
// ---------------------------------------------------------------------------
__global__ void zero_u32_kernel(unsigned* p, int n) {
  for (int i = blockIdx.x * blockDim.x + threadIdx.x; i < n; i += gridDim.x * blockDim.x)
    p[i] = 0u;
}

// Convert fp32 [N,Ks] row-major into bf16 [N,Kdst] at column offset colofs.
__global__ void pack_w_kernel(const float* __restrict__ src, __hip_bfloat16* __restrict__ dst,
                              int N, int Ks, int Kdst, int colofs) {
  const int total = N * Ks;
  for (int i = blockIdx.x * blockDim.x + threadIdx.x; i < total; i += gridDim.x * blockDim.x) {
    const int r = i / Ks, k = i - r * Ks;
    dst[(size_t)r * Kdst + colofs + k] = __float2bfloat16(src[i]);
  }
}

// ---------------------------------------------------------------------------
// Persistent encoder pass: one LSTM layer, one direction, 256 steps.
// This block's 64 gate-weight rows (4 gates x 16 h-cols) are staged once into
// LDS via async-to-LDS and served from ds_load for all 256 steps.
// ---------------------------------------------------------------------------
__global__ void enc_pass_kernel(
    const int* __restrict__ source, const __hip_bfloat16* __restrict__ embb,
    const __hip_bfloat16* __restrict__ y0,
    const __hip_bfloat16* __restrict__ Wcat,
    const float* __restrict__ bih, const float* __restrict__ bhh,
    float* __restrict__ h, float* __restrict__ c,
    __hip_bfloat16* __restrict__ xh,
    __hip_bfloat16* __restrict__ yout, long long ytstride, long long ybstride, int ycolofs,
    int Kin, int reverse, unsigned* bar) {
  const int K = Kin + H_;
  const int KP = K + LPAD;
  const int nth = gridDim.x * blockDim.x;
  const int gtid = blockIdx.x * blockDim.x + threadIdx.x;
  const int j0 = blockIdx.x * 16;
  const int row0 = (threadIdx.x >> 5) * 16;
  unsigned barc = 0;

  extern __shared__ __align__(16) char dynsm[];
  uint16_t* Wl = (uint16_t*)dynsm;                 // [64][KP] gate-rows tile

  // Stage weight tile: LDS row g*16+jr  <-  Wcat row g*512 + j0 + jr
  {
    const int kc8 = K >> 3;                        // 16B chunks per row
    for (int chk = threadIdx.x; chk < 64 * kc8; chk += NT) {
      const int row = chk / kc8;
      const int kc  = (chk - row * kc8) << 3;
      const int g = row >> 4, jr = row & 15;
      const uint16_t* ga = (const uint16_t*)Wcat + (size_t)(g * 512 + j0 + jr) * K + kc;
      async_copy_b128(lds_addr_of(Wl + row * KP + kc), ga);
    }
    wait_asynccnt0();
  }
  __syncthreads();

  // zero initial state
  for (int i = gtid; i < B_ * H_; i += nth) { h[i] = 0.0f; c[i] = 0.0f; }
  gsync(bar, barc, gridDim.x);

  for (int t = 0; t < S_; ++t) {
    const int tt = reverse ? (S_ - 1 - t) : t;

    // Phase A: build bf16 activation rows  xh[b, 0:K] = [x_t(b) | h(b)]
    const int total = B_ * K;
    for (int i = gtid; i < total; i += nth) {
      const int b = i / K, k = i - b * K;
      __hip_bfloat16 v;
      if (k >= Kin) {
        v = __float2bfloat16(h[b * H_ + (k - Kin)]);
      } else if (Kin == E_) {            // layer 0: embedding lookup (padding_idx=0)
        const int tok = source[b * S_ + tt];
        v = tok ? embb[tok * E_ + k] : __float2bfloat16(0.0f);
      } else {                           // layer 1: concat bidirectional layer-0 output
        v = y0[(size_t)tt * (B_ * 2 * H_) + (size_t)b * (2 * H_) + k];
      }
      xh[i] = v;
    }
    gsync(bar, barc, gridDim.x);

    // Phase B: gates GEMM (A from global, W from LDS) + LSTM cell
    const long long obase = ytstride * (long long)tt + (long long)ycolofs;
    lstm_cell_tile((const uint16_t*)xh, K, row0,
                   Wl, KP, 0, 16, 32, 48, j0, K,
                   bih, bhh, h, c,
                   yout, obase, ybstride);
    gsync(bar, barc, gridDim.x);
  }
}

// ---------------------------------------------------------------------------
// Persistent decoder: attention + 2 LSTM cells + vocab projection, 256 steps.
// LDS holds this block's W0 gate tile (64x1792), Wq tile (32x512) and Wout
// tile (16x512); W1 stays in global (L2-resident).
// ---------------------------------------------------------------------------
__global__ void dec_kernel(
    const int* __restrict__ source, const int* __restrict__ dtoks,
    const __hip_bfloat16* __restrict__ embb, const __hip_bfloat16* __restrict__ enc_out,
    const __hip_bfloat16* __restrict__ Wqb, const float* __restrict__ bq,
    const __hip_bfloat16* __restrict__ W0, const float* __restrict__ b0i, const float* __restrict__ b0h,
    const __hip_bfloat16* __restrict__ W1, const float* __restrict__ b1i, const float* __restrict__ b1h,
    const __hip_bfloat16* __restrict__ Woutb, const float* __restrict__ bout,
    const float* __restrict__ hf0, const float* __restrict__ cf0,
    const float* __restrict__ hb0, const float* __restrict__ cb0,
    const float* __restrict__ hf1, const float* __restrict__ cf1,
    const float* __restrict__ hb1, const float* __restrict__ cb1,
    float* __restrict__ d0h, float* __restrict__ d0c,
    float* __restrict__ d1h, float* __restrict__ d1c, __hip_bfloat16* __restrict__ d1hb,
    float* __restrict__ q, float* __restrict__ attn,
    __hip_bfloat16* __restrict__ xh0, __hip_bfloat16* __restrict__ xh1,
    float* __restrict__ out, unsigned* bar) {
  const int nth = gridDim.x * blockDim.x;
  const int gtid = blockIdx.x * blockDim.x + threadIdx.x;
  const int lane = threadIdx.x & 31;
  const int j0 = blockIdx.x * 16;
  const int row0 = (threadIdx.x >> 5) * 16;
  unsigned barc = 0;

  const int K0 = E_ + 2 * H_ + H_;  // 1792
  const int K1 = 2 * H_;            // 1024
  const int K0P = K0 + LPAD;        // 1800
  const int HP  = H_ + LPAD;        // 520

  extern __shared__ __align__(16) char dynsm[];
  uint16_t* W0l   = (uint16_t*)dynsm;              // [64][K0P]
  uint16_t* Wql   = W0l + 64 * K0P;                // [32][HP]
  uint16_t* Woutl = Wql + 32 * HP;                 // [16][HP]

  // Stage LDS weight tiles (once for all 256 steps)
  {
    const int kc8 = K0 >> 3;
    for (int chk = threadIdx.x; chk < 64 * kc8; chk += NT) {
      const int row = chk / kc8;
      const int kc  = (chk - row * kc8) << 3;
      const int g = row >> 4, jr = row & 15;
      const uint16_t* ga = (const uint16_t*)W0 + (size_t)(g * 512 + j0 + jr) * K0 + kc;
      async_copy_b128(lds_addr_of(W0l + row * K0P + kc), ga);
    }
    const int hc8 = H_ >> 3;
    for (int chk = threadIdx.x; chk < 32 * hc8; chk += NT) {
      const int row = chk / hc8;
      const int kc  = (chk - row * hc8) << 3;
      const uint16_t* ga = (const uint16_t*)Wqb + (size_t)(blockIdx.x * 32 + row) * H_ + kc;
      async_copy_b128(lds_addr_of(Wql + row * HP + kc), ga);
    }
    for (int chk = threadIdx.x; chk < 16 * hc8; chk += NT) {
      const int row = chk / hc8;
      const int kc  = (chk - row * hc8) << 3;
      const uint16_t* ga = (const uint16_t*)Woutb + (size_t)(blockIdx.x * 16 + row) * H_ + kc;
      async_copy_b128(lds_addr_of(Woutl + row * HP + kc), ga);
    }
    wait_asynccnt0();
  }
  __syncthreads();

  // Init: merge bidirectional encoder final states (sum over directions)
  for (int i = gtid; i < B_ * H_; i += nth) {
    d0h[i] = hf0[i] + hb0[i];
    d0c[i] = cf0[i] + cb0[i];
    const float h1 = hf1[i] + hb1[i];
    d1h[i] = h1;
    d1c[i] = cf1[i] + cb1[i];
    d1hb[i] = __float2bfloat16(h1);
  }
  gsync(bar, barc, gridDim.x);

  for (int t = 0; t < T_; ++t) {
    // P1: attention query q = d1h @ Wq^T + bq   [128 x 1024]  (Wq from LDS)
    for (int c0 = 0; c0 < 32; c0 += 16)
      gemm_out_tile((const uint16_t*)d1hb, H_, row0,
                    Wql, HP, c0, blockIdx.x * 32 + c0, H_, bq, q, 0, 2 * H_);
    gsync(bar, barc, gridDim.x);

    // P2: scores[b,s] = <enc_out[b,s,:], q[b,:]> with padding mask (16B loads)
    for (int i = gtid; i < B_ * S_; i += nth) {
      const int b = i >> 8, s = i & (S_ - 1);
      const uint16_t* eb = (const uint16_t*)enc_out + ((size_t)b * S_ + s) * (2 * H_);
      const float* qb = q + b * (2 * H_);
      float acc = 0.0f;
      for (int k = 0; k < 2 * H_; k += 8) {
        uint4 raw = *reinterpret_cast<const uint4*>(eb + k);
        const __hip_bfloat16* hv = (const __hip_bfloat16*)&raw;
        #pragma unroll
        for (int j = 0; j < 8; ++j) acc += __bfloat162float(hv[j]) * qb[k + j];
      }
      attn[i] = (source[b * S_ + s] == 0) ? -1e9f : acc;
    }
    gsync(bar, barc, gridDim.x);

    // P3: softmax over S per batch row (one wave per b)
    {
      const int w = gtid >> 5;
      if (w < B_) {
        const int b = w;
        float v[8], mx = -3.4e38f;
        #pragma unroll
        for (int r = 0; r < 8; ++r) { v[r] = attn[b * S_ + lane + 32 * r]; mx = fmaxf(mx, v[r]); }
        for (int off = 16; off; off >>= 1) mx = fmaxf(mx, __shfl_xor(mx, off));
        float sum = 0.0f;
        #pragma unroll
        for (int r = 0; r < 8; ++r) { v[r] = __expf(v[r] - mx); sum += v[r]; }
        for (int off = 16; off; off >>= 1) sum += __shfl_xor(sum, off);
        const float inv = 1.0f / sum;
        #pragma unroll
        for (int r = 0; r < 8; ++r) attn[b * S_ + lane + 32 * r] = v[r] * inv;
      }
    }
    gsync(bar, barc, gridDim.x);

    // P4a: embedding part of xh0
    for (int i = gtid; i < B_ * E_; i += nth) {
      const int b = i >> 8, k = i & (E_ - 1);
      const int tok = dtoks[b * T_ + t];
      xh0[b * K0 + k] = tok ? embb[tok * E_ + k] : __float2bfloat16(0.0f);
    }
    // P4b: ctx = attn @ enc_out  (each thread: 8 contiguous cols, 16B loads)
    for (int i = gtid; i < B_ * (2 * H_ / 8); i += nth) {
      const int b = i >> 7;
      const int c0 = (i & 127) << 3;
      const float* ab = attn + b * S_;
      const uint16_t* eb = (const uint16_t*)enc_out + (size_t)b * S_ * (2 * H_) + c0;
      float acc[8] = {0.f, 0.f, 0.f, 0.f, 0.f, 0.f, 0.f, 0.f};
      for (int s = 0; s < S_; ++s) {
        uint4 raw = *reinterpret_cast<const uint4*>(eb + (size_t)s * (2 * H_));
        const __hip_bfloat16* hv = (const __hip_bfloat16*)&raw;
        const float a = ab[s];
        #pragma unroll
        for (int j = 0; j < 8; ++j) acc[j] += a * __bfloat162float(hv[j]);
      }
      #pragma unroll
      for (int j = 0; j < 8; ++j) xh0[b * K0 + E_ + c0 + j] = __float2bfloat16(acc[j]);
    }
    // P4c: recurrent halves: xh0 tail from d0h, xh1 tail from d1h
    for (int i = gtid; i < B_ * H_; i += nth) {
      const int b = i >> 9, j = i & (H_ - 1);
      xh0[b * K0 + (E_ + 2 * H_) + j] = __float2bfloat16(d0h[i]);
      xh1[b * K1 + H_ + j]            = __float2bfloat16(d1h[i]);
    }
    gsync(bar, barc, gridDim.x);

    // P5: decoder cell 0 (K=1792, W0 from LDS); h mirrors into xh1[:, 0:512]
    lstm_cell_tile((const uint16_t*)xh0, K0, row0,
                   W0l, K0P, 0, 16, 32, 48, j0, K0,
                   b0i, b0h, d0h, d0c, xh1, 0, K1);
    gsync(bar, barc, gridDim.x);

    // P6: decoder cell 1 (K=1024, W1 from global/L2); h mirrors into d1hb
    lstm_cell_tile((const uint16_t*)xh1, K1, row0,
                   (const uint16_t*)W1, K1, j0, 512 + j0, 1024 + j0, 1536 + j0, j0, K1,
                   b1i, b1h, d1h, d1c, d1hb, 0, H_);
    gsync(bar, barc, gridDim.x);

    // P7: logits = d1h @ Wout^T + bout -> out[b, t, :]  (Wout from LDS)
    gemm_out_tile((const uint16_t*)d1hb, H_, row0,
                  Woutl, HP, 0, blockIdx.x * 16, H_,
                  bout, out, (long long)t * V_, (long long)T_ * V_);
    gsync(bar, barc, gridDim.x);
  }
}

// ---------------------------------------------------------------------------
// Host launcher
// ---------------------------------------------------------------------------
extern "C" void kernel_launch(void* const* d_in, const int* in_sizes, int n_in,
                              void* d_out, int out_size, void* d_ws, size_t ws_size,
                              hipStream_t stream) {
  (void)in_sizes; (void)n_in; (void)out_size; (void)ws_size;
  const int*   source = (const int*)d_in[0];
  const int*   dtoks  = (const int*)d_in[1];
  const float* emb    = (const float*)d_in[2];
  const float* eWih0  = (const float*)d_in[3];
  const float* eWhh0  = (const float*)d_in[4];
  const float* ebih0  = (const float*)d_in[5];
  const float* ebhh0  = (const float*)d_in[6];
  const float* eWih1  = (const float*)d_in[7];
  const float* eWhh1  = (const float*)d_in[8];
  const float* ebih1  = (const float*)d_in[9];
  const float* ebhh1  = (const float*)d_in[10];
  const float* dWih0  = (const float*)d_in[11];
  const float* dWhh0  = (const float*)d_in[12];
  const float* dbih0  = (const float*)d_in[13];
  const float* dbhh0  = (const float*)d_in[14];
  const float* dWih1  = (const float*)d_in[15];
  const float* dWhh1  = (const float*)d_in[16];
  const float* dbih1  = (const float*)d_in[17];
  const float* dbhh1  = (const float*)d_in[18];
  const float* Wq     = (const float*)d_in[19];
  const float* bq     = (const float*)d_in[20];
  const float* Wout   = (const float*)d_in[21];
  const float* bout   = (const float*)d_in[22];
  float* out = (float*)d_out;

  // ---- carve workspace (256B aligned) ----
  char* w = (char*)d_ws;
  auto alloc = [&](size_t bytes) -> void* {
    void* p = (void*)w;
    w += (bytes + 255) & ~(size_t)255;
    return p;
  };
  unsigned*        sync  = (unsigned*)alloc(64 * sizeof(unsigned));
  __hip_bfloat16*  embb  = (__hip_bfloat16*)alloc((size_t)V_ * E_ * 2);
  __hip_bfloat16*  We0f  = (__hip_bfloat16*)alloc((size_t)2048 * 768 * 2);
  __hip_bfloat16*  We0b  = (__hip_bfloat16*)alloc((size_t)2048 * 768 * 2);
  __hip_bfloat16*  We1f  = (__hip_bfloat16*)alloc((size_t)2048 * 1536 * 2);
  __hip_bfloat16*  We1b  = (__hip_bfloat16*)alloc((size_t)2048 * 1536 * 2);
  __hip_bfloat16*  Wd0   = (__hip_bfloat16*)alloc((size_t)2048 * 1792 * 2);
  __hip_bfloat16*  Wd1   = (__hip_bfloat16*)alloc((size_t)2048 * 1024 * 2);
  __hip_bfloat16*  Wqb   = (__hip_bfloat16*)alloc((size_t)1024 * 512 * 2);
  __hip_bfloat16*  Woutb = (__hip_bfloat16*)alloc((size_t)512 * 512 * 2);
  __hip_bfloat16*  y0    = (__hip_bfloat16*)alloc((size_t)S_ * B_ * 1024 * 2);     // [S][B][2H]
  __hip_bfloat16*  encO  = (__hip_bfloat16*)alloc((size_t)B_ * S_ * 1024 * 2);     // [B][S][2H]
  __hip_bfloat16*  xhA   = (__hip_bfloat16*)alloc((size_t)B_ * 1536 * 2);
  __hip_bfloat16*  xhB   = (__hip_bfloat16*)alloc((size_t)B_ * 1536 * 2);
  __hip_bfloat16*  xhC   = (__hip_bfloat16*)alloc((size_t)B_ * 1536 * 2);
  __hip_bfloat16*  xhD   = (__hip_bfloat16*)alloc((size_t)B_ * 1536 * 2);
  float* hf0 = (float*)alloc((size_t)B_ * H_ * 4);
  float* cf0 = (float*)alloc((size_t)B_ * H_ * 4);
  float* hb0 = (float*)alloc((size_t)B_ * H_ * 4);
  float* cb0 = (float*)alloc((size_t)B_ * H_ * 4);
  float* hf1 = (float*)alloc((size_t)B_ * H_ * 4);
  float* cf1 = (float*)alloc((size_t)B_ * H_ * 4);
  float* hb1 = (float*)alloc((size_t)B_ * H_ * 4);
  float* cb1 = (float*)alloc((size_t)B_ * H_ * 4);
  float* d0h = (float*)alloc((size_t)B_ * H_ * 4);
  float* d0c = (float*)alloc((size_t)B_ * H_ * 4);
  float* d1h = (float*)alloc((size_t)B_ * H_ * 4);
  float* d1c = (float*)alloc((size_t)B_ * H_ * 4);
  __hip_bfloat16* d1hb = (__hip_bfloat16*)alloc((size_t)B_ * H_ * 2);
  float* qbuf = (float*)alloc((size_t)B_ * 1024 * 4);
  float* attn = (float*)alloc((size_t)B_ * S_ * 4);
  __hip_bfloat16* xh0 = (__hip_bfloat16*)alloc((size_t)B_ * 1792 * 2);
  __hip_bfloat16* xh1 = (__hip_bfloat16*)alloc((size_t)B_ * 1024 * 2);

  // ---- init sync counters ----
  zero_u32_kernel<<<1, 64, 0, stream>>>(sync, 64);

  // ---- pack weights fp32 -> bf16 (concat [Wih ; Whh] along K) ----
  auto pack = [&](const float* src, __hip_bfloat16* dst, int N, int Ks, int Kd, int ofs) {
    int blocks = (N * Ks + NT - 1) / NT;
    if (blocks > 512) blocks = 512;
    pack_w_kernel<<<blocks, NT, 0, stream>>>(src, dst, N, Ks, Kd, ofs);
  };
  pack(emb, embb, V_, E_, E_, 0);
  pack(eWih0,              We0f, 2048, 256, 768, 0);
  pack(eWhh0,              We0f, 2048, 512, 768, 256);
  pack(eWih0 + 2048 * 256, We0b, 2048, 256, 768, 0);
  pack(eWhh0 + 2048 * 512, We0b, 2048, 512, 768, 256);
  pack(eWih1,               We1f, 2048, 1024, 1536, 0);
  pack(eWhh1,               We1f, 2048, 512,  1536, 1024);
  pack(eWih1 + 2048 * 1024, We1b, 2048, 1024, 1536, 0);
  pack(eWhh1 + 2048 * 512,  We1b, 2048, 512,  1536, 1024);
  pack(dWih0, Wd0, 2048, 1280, 1792, 0);
  pack(dWhh0, Wd0, 2048, 512,  1792, 1280);
  pack(dWih1, Wd1, 2048, 512, 1024, 0);
  pack(dWhh1, Wd1, 2048, 512, 1024, 512);
  pack(Wq,   Wqb,   1024, 512, 512, 0);
  pack(Wout, Woutb, 512,  512, 512, 0);

  // ---- dynamic LDS sizes (padded weight tiles) ----
  const size_t smem_enc0 = (size_t)64 * (768 + LPAD)  * 2;                       //  ~97 KB
  const size_t smem_enc1 = (size_t)64 * (1536 + LPAD) * 2;                       // ~193 KB
  const size_t smem_dec  = ((size_t)64 * (1792 + LPAD)
                          + (size_t)32 * (512 + LPAD)
                          + (size_t)16 * (512 + LPAD)) * 2;                      // ~274 KB

  // ---- encoder: 4 persistent layer-direction passes ----
  enc_pass_kernel<<<NB, NT, smem_enc0, stream>>>(source, embb, y0, We0f, ebih0, ebhh0,
      hf0, cf0, xhA, y0, (long long)(B_ * 1024), 1024LL, 0,   E_,  0, sync + 0);
  enc_pass_kernel<<<NB, NT, smem_enc0, stream>>>(source, embb, y0, We0b, ebih0 + 2048, ebhh0 + 2048,
      hb0, cb0, xhB, y0, (long long)(B_ * 1024), 1024LL, 512, E_,  1, sync + 1);
  enc_pass_kernel<<<NB, NT, smem_enc1, stream>>>(source, embb, y0, We1f, ebih1, ebhh1,
      hf1, cf1, xhC, encO, 1024LL, (long long)(S_ * 1024), 0,   2 * H_, 0, sync + 2);
  enc_pass_kernel<<<NB, NT, smem_enc1, stream>>>(source, embb, y0, We1b, ebih1 + 2048, ebhh1 + 2048,
      hb1, cb1, xhD, encO, 1024LL, (long long)(S_ * 1024), 512, 2 * H_, 1, sync + 3);

  // ---- decoder: persistent attention + 2-cell + projection loop ----
  dec_kernel<<<NB, NT, smem_dec, stream>>>(source, dtoks, embb, encO,
      Wqb, bq, Wd0, dbih0, dbhh0, Wd1, dbih1, dbhh1, Woutb, bout,
      hf0, cf0, hb0, cb0, hf1, cf1, hb1, cb1,
      d0h, d0c, d1h, d1c, d1hb, qbuf, attn, xh0, xh1,
      out, sync + 4);
}
